// RelationalAttention_33397665694638
// MI455X (gfx1250) — matrix operand order repaired and checked by reading, not compiled, not run on hardware
//
#include <hip/hip_runtime.h>
#include <hip/hip_bf16.h>

// ---------------------------------------------------------------------------
// CDNA5 (gfx1250) wave32 WMMA implementation of relational attention block.
// B=2, N=384, D=128.  All matmuls via v_wmma_f32_16x16x32_bf16 (f32 accum).
// 32-row tiles, register-prefetch software pipeline, global_prefetch for t+2.
// ---------------------------------------------------------------------------

#define BATCH 2
#define NN    384
#define DD    128
#define BN    (BATCH * NN)     // 768

typedef __attribute__((ext_vector_type(16))) __bf16 v16bf;
typedef __attribute__((ext_vector_type(8)))  float  v8f;

union FragAB { v16bf v; unsigned int u[8]; };

__device__ __forceinline__ unsigned short f2bf(float f) {
  unsigned int u = __float_as_uint(f);
  u += 0x7FFFu + ((u >> 16) & 1u);          // round-to-nearest-even
  return (unsigned short)(u >> 16);
}
__device__ __forceinline__ unsigned int pack2(float lo, float hi) {
  return (unsigned int)f2bf(lo) | ((unsigned int)f2bf(hi) << 16);
}

// A fragment: 16x32 bf16, row-major LDS (stride in bf16 elems).
// ISA 7.12.2: lanes 0-15 row M=lane, K={0..7,16..23}; lanes 16-31 K={8..15,24..31}
__device__ __forceinline__ v16bf loadFragA(const unsigned short* base, int rowBase,
                                           int stride, int k0, int lane) {
  FragAB f;
  const unsigned short* p = base + (size_t)(rowBase + (lane & 15)) * stride;
  int kb = k0 + ((lane >> 4) << 3);
#pragma unroll
  for (int r = 0; r < 8; ++r) {
    int k = kb + ((r >> 2) << 4) + ((r & 3) << 1);
    f.u[r] = *(const unsigned int*)(p + k);
  }
  return f.v;
}
// B fragment: 32x16 bf16 = W^T where W (out,in) row-major in LDS. Lane selects
// output column (a W row); lanes 0-15 K={0..7,16..23}, lanes 16-31 K={8..15,24..31}.
__device__ __forceinline__ v16bf loadFragB(const unsigned short* base, int rowBase,
                                           int stride, int k0, int lane) {
  FragAB f;
  const unsigned short* p = base + (size_t)(rowBase + (lane & 15)) * stride;
  int kb = k0 + ((lane >> 4) << 3);
#pragma unroll
  for (int r = 0; r < 8; ++r)
    f.u[r] = *(const unsigned int*)(p + kb + ((r >> 2) << 4) + ((r & 3) << 1));
  return f.v;
}

// acc += A(16xK) * W(16 out rows, K in)^T  via WMMA, K multiple of 32
__device__ __forceinline__ v8f gemm16(const unsigned short* A, int aRow, int aStride,
                                      const unsigned short* W, int wRow, int wStride,
                                      int K, v8f acc, int lane) {
  for (int k0 = 0; k0 < K; k0 += 32) {
    v16bf a = loadFragA(A, aRow, aStride, k0, lane);
    v16bf b = loadFragB(W, wRow, wStride, k0, lane);
    acc = __builtin_amdgcn_wmma_f32_16x16x32_bf16(false, a, false, b,
                                                  (short)0, acc, false, false);
  }
  return acc;
}

// ---------------------------------------------------------------------------
// Kernel 1/3: out[row, :] = A[row, :] @ W[:, colOff:colOff+128]^T (+bias)
// A: (768,128) f32 contiguous.  W row-major (128, wStride).  grid.x = 48.
// ---------------------------------------------------------------------------
__global__ __launch_bounds__(256) void k_lin128(const float* __restrict__ A,
                                                const float* __restrict__ w,
                                                int wStride, int wColOff,
                                                const float* __restrict__ bias,
                                                float* __restrict__ out) {
  __shared__ __align__(16) unsigned short sA[16 * 128];
  __shared__ __align__(16) unsigned short sW[128 * 128];
  const int t = threadIdx.x;
  const int rowBase = blockIdx.x * 16;
  for (int p = t; p < 128 * 64; p += 256) {
    int o = p >> 6, pr = (p & 63) << 1;
    float2 v = *(const float2*)(w + (size_t)o * wStride + wColOff + pr);
    ((unsigned int*)sW)[p] = pack2(v.x, v.y);
  }
  for (int p = t; p < 16 * 64; p += 256) {
    float2 v = ((const float2*)(A + (size_t)rowBase * DD))[p];
    ((unsigned int*)sA)[p] = pack2(v.x, v.y);
  }
  __syncthreads();
  const int lane = t & 31, wv = t >> 5;
  const int ob = wv * 16;
  const int ln = lane & 15, lh = lane >> 4;
  v8f acc = {};
  acc = gemm16(sA, 0, 128, sW, ob, 128, 128, acc, lane);
  const int n = ob + ln;
  const float bv = bias ? bias[n] : 0.0f;
#pragma unroll
  for (int r = 0; r < 8; ++r)
    out[(size_t)(rowBase + r + lh * 8) * DD + n] = acc[r] + bv;
}

// ---------------------------------------------------------------------------
// Kernel 2: edge-conditioned attention.  One block per (b,n).  Two passes over
// ef[b,n,:,:] in 32-row tiles: (1) q,k -> scores, softmax; (2) recompute v,
// z-weighted sum accumulated in fragment registers.  ef is L2-resident (151MB
// < 192MB L2) so the re-read is cheap.  Loads pipelined through registers.
// ---------------------------------------------------------------------------
__global__ __launch_bounds__(256) void k_attn(const float* __restrict__ ef,
    const float* __restrict__ wq, const float* __restrict__ bq,
    const float* __restrict__ wk, const float* __restrict__ bk,
    const float* __restrict__ wv, const float* __restrict__ bv,
    const float* __restrict__ qn, const float* __restrict__ kn,
    const float* __restrict__ vn,
    float* __restrict__ xnOut, float* __restrict__ xnWs) {
  __shared__ __align__(16) unsigned short sW[3 * 128 * 128];  // 96 KB bf16
  __shared__ __align__(16) unsigned short sEf[32 * 128];      //  8 KB
  __shared__ float sQ[32 * 128];                              // 16 KB
  __shared__ float sK[32 * 128];                              // 16 KB
  __shared__ float sNode[3 * 128];
  __shared__ float sS[NN];
  __shared__ float sRed[16];
  const int t = threadIdx.x;
  const int bn = blockIdx.x;
  // stage weights (f32 -> packed bf16)
  for (int g = 0; g < 3; ++g) {
    const float* w = (g == 0) ? wq : (g == 1) ? wk : wv;
    for (int p = t; p < 128 * 64; p += 256) {
      float2 v = ((const float2*)w)[p];
      ((unsigned int*)sW)[g * 128 * 64 + p] = pack2(v.x, v.y);
    }
  }
  if (t < 128) {
    sNode[t]       = qn[(size_t)bn * DD + t] + bq[t];
    sNode[128 + t] = kn[(size_t)bn * DD + t] + bk[t];
    sNode[256 + t] = vn[(size_t)bn * DD + t] + bv[t];
  }
  const int lane = t & 31, wvv = t >> 5, ob = wvv * 16;
  const int ln = lane & 15, lh = lane >> 4;
  const int n = ob + ln;

  float2 rij[8];
  auto loadTile = [&](int mt) {
    const float2* src = (const float2*)(ef + ((size_t)bn * NN + mt * 32) * DD);
#pragma unroll
    for (int i2 = 0; i2 < 8; ++i2) rij[i2] = src[t + 256 * i2];
  };
  auto packTile = [&]() {
#pragma unroll
    for (int i2 = 0; i2 < 8; ++i2)
      ((unsigned int*)sEf)[t + 256 * i2] = pack2(rij[i2].x, rij[i2].y);
  };

  // ---- pass 1: scores ----
  loadTile(0);
  for (int mt = 0; mt < 12; ++mt) {
    __syncthreads();
    packTile();
    __syncthreads();
    if (mt + 1 < 12) loadTile(mt + 1);                 // pipeline next tile
    if (mt + 2 < 12)                                   // L2 prefetch for t+2
      __builtin_prefetch(ef + ((size_t)bn * NN + (mt + 2) * 32) * DD + t * 16, 0, 3);
#pragma unroll
    for (int rt = 0; rt < 2; ++rt) {
      v8f aq = {}; aq = gemm16(sEf, rt * 16, 128, sW,         ob, 128, 128, aq, lane);
      v8f ak = {}; ak = gemm16(sEf, rt * 16, 128, sW + 16384, ob, 128, 128, ak, lane);
      float nq = sNode[n], nk = sNode[128 + n];
#pragma unroll
      for (int r = 0; r < 8; ++r) {
        int m = rt * 16 + r + lh * 8;
        sQ[m * 128 + n] = aq[r] + nq;
        sK[m * 128 + n] = ak[r] + nk;
      }
    }
    __syncthreads();
    {   // row dots: 8 threads per row, 16 cols each
      int row = t >> 3, cb = (t & 7) << 4;
      float s = 0.f;
#pragma unroll
      for (int c = 0; c < 16; ++c) s += sQ[row * 128 + cb + c] * sK[row * 128 + cb + c];
      s += __shfl_xor(s, 1); s += __shfl_xor(s, 2); s += __shfl_xor(s, 4);
      if ((t & 7) == 0) sS[mt * 32 + row] = s * 0.088388347648318447f;  // /sqrt(128)
    }
  }
  __syncthreads();
  // ---- softmax over 384 ----
  float mx = -3.4e38f;
  if (t < 128) mx = fmaxf(sS[t], fmaxf(sS[t + 128], sS[t + 256]));
  for (int off = 16; off >= 1; off >>= 1) mx = fmaxf(mx, __shfl_xor(mx, off));
  if (lane == 0) sRed[wvv] = mx;
  __syncthreads();
  if (t == 0) {
    float m0 = sRed[0];
    for (int i = 1; i < 8; ++i) m0 = fmaxf(m0, sRed[i]);
    sRed[8] = m0;
  }
  __syncthreads();
  const float M = sRed[8];
  float ssum = 0.f;
  if (t < 128) {
    for (int j = t; j < NN; j += 128) {
      float e = __expf(sS[j] - M);
      sS[j] = e; ssum += e;
    }
  }
  for (int off = 16; off >= 1; off >>= 1) ssum += __shfl_xor(ssum, off);
  if (lane == 0) sRed[wvv] = ssum;
  __syncthreads();
  if (t == 0) {
    float s = 0.f;
    for (int i = 0; i < 8; ++i) s += sRed[i];
    sRed[9] = 1.0f / s;
  }
  __syncthreads();
  const float inv = sRed[9];

  // ---- pass 2: recompute v tiles (L2 hits), accumulate xn in registers ----
  float xacc = 0.f;
  loadTile(0);
  for (int mt = 0; mt < 12; ++mt) {
    __syncthreads();
    packTile();
    __syncthreads();
    if (mt + 1 < 12) loadTile(mt + 1);
#pragma unroll
    for (int rt = 0; rt < 2; ++rt) {
      v8f av = {}; av = gemm16(sEf, rt * 16, 128, sW + 32768, ob, 128, 128, av, lane);
#pragma unroll
      for (int r = 0; r < 8; ++r)
        xacc += sS[mt * 32 + rt * 16 + r + lh * 8] * av[r];
    }
  }
  xacc += __shfl_xor(xacc, 16);          // combine row halves (same column n)
  if (lh == 0) {
    float res = xacc * inv + sNode[256 + n];   // +vn (+biases), since sum z == 1
    xnOut[(size_t)bn * DD + n] = res;
    xnWs [(size_t)bn * DD + n] = res;
  }
}

// ---------------------------------------------------------------------------
// Kernel 4: edge MLP.  One block per (b,i); loop over 12 j-tiles of 32 rows.
// 5 weight blocks resident as bf16 in LDS (160 KB); total ~232 KB of 320 KB.
// Each wave runs two independent WMMA chains per GEMM (row halves).
// ---------------------------------------------------------------------------
__global__ __launch_bounds__(256) void k_edge(const float* __restrict__ ef,
    const float* __restrict__ mw,  const float* __restrict__ mb,
    const float* __restrict__ w5,  const float* __restrict__ b5,
    const float* __restrict__ g1,  const float* __restrict__ bb1,
    const float* __restrict__ e1w, const float* __restrict__ e1b,
    const float* __restrict__ e2w, const float* __restrict__ e2b,
    const float* __restrict__ g2,  const float* __restrict__ bb2,
    const float* __restrict__ c1,  const float* __restrict__ c2,
    float* __restrict__ eOut) {
  __shared__ __align__(16) unsigned short sWm3[128 * 128];   // m_w[:,256:384]
  __shared__ __align__(16) unsigned short sWm4[128 * 128];   // m_w[:,384:512]
  __shared__ __align__(16) unsigned short sW5 [128 * 128];
  __shared__ __align__(16) unsigned short sE1 [128 * 128];
  __shared__ __align__(16) unsigned short sE2 [128 * 128];
  __shared__ __align__(16) unsigned short sEij[32 * 128];
  __shared__ __align__(16) unsigned short sEji[32 * 128];
  __shared__ __align__(16) unsigned short sM  [32 * 128];
  __shared__ __align__(16) unsigned short sUb [32 * 128];
  __shared__ __align__(16) unsigned short sR  [32 * 128];
  __shared__ float sEF[32 * 128];   // f32 ef_ij, reused as GEMM4 output
  __shared__ float sU [32 * 128];   // LayerNorm-1 output (f32)
  __shared__ float sC1[128];
  const int t = threadIdx.x;
  const int bi = blockIdx.x;                  // b*384 + i
  const int b = bi / NN, i = bi % NN;
  for (int p = t; p < 128 * 64; p += 256) {
    int o = p >> 6, pr = (p & 63) << 1;
    float2 v3 = *(const float2*)(mw + (size_t)o * 512 + 256 + pr);
    float2 v4 = *(const float2*)(mw + (size_t)o * 512 + 384 + pr);
    ((unsigned int*)sWm3)[p] = pack2(v3.x, v3.y);
    ((unsigned int*)sWm4)[p] = pack2(v4.x, v4.y);
    float2 v5 = ((const float2*)w5)[p];  ((unsigned int*)sW5)[p] = pack2(v5.x, v5.y);
    float2 va = ((const float2*)e1w)[p]; ((unsigned int*)sE1)[p] = pack2(va.x, va.y);
    float2 vb = ((const float2*)e2w)[p]; ((unsigned int*)sE2)[p] = pack2(vb.x, vb.y);
  }
  if (t < 128) sC1[t] = c1[(size_t)bi * DD + t] + mb[t];
  const int lane = t & 31, wvv = t >> 5, ob = wvv * 16;
  const int ln = lane & 15, lh = lane >> 4;
  const int n = ob + ln;

  float2 rij[8], rji[8];
  auto loadTiles = [&](int jt) {
    const int jb = jt * 32;
    const float2* src = (const float2*)(ef + ((size_t)bi * NN + jb) * DD);
#pragma unroll
    for (int i2 = 0; i2 < 8; ++i2) {
      int p = t + 256 * i2;
      rij[i2] = src[p];
      int r = p >> 6, pr = (p & 63) << 1;       // 64 float2 per 128-wide row
      rji[i2] = *(const float2*)(ef + ((size_t)(b * NN + jb + r) * NN + i) * DD + pr);
    }
  };
  loadTiles(0);

  for (int jt = 0; jt < 12; ++jt) {
    __syncthreads();
    const int jb = jt * 32;
#pragma unroll
    for (int i2 = 0; i2 < 8; ++i2) {          // pack current tile into LDS
      int p = t + 256 * i2;
      ((unsigned int*)sEij)[p] = pack2(rij[i2].x, rij[i2].y);
      sEF[2 * p] = rij[i2].x; sEF[2 * p + 1] = rij[i2].y;
      ((unsigned int*)sEji)[p] = pack2(rji[i2].x, rji[i2].y);
    }
    __syncthreads();
    if (jt + 1 < 12) loadTiles(jt + 1);       // pipeline next tile's loads
    if (jt + 2 < 12) {                        // L2 prefetch two tiles ahead
      const int jb2 = (jt + 2) * 32;
      __builtin_prefetch(ef + ((size_t)bi * NN + jb2) * DD + t * 16, 0, 3);
      __builtin_prefetch(ef + ((size_t)(b * NN + jb2 + (t >> 3)) * NN + i) * DD
                            + (t & 7) * 16, 0, 3);
    }
    // GEMM1: m = relu(c1_i + c2_j + ef_ij@Wm3^T + ef_ji@Wm4^T + m_b)
#pragma unroll
    for (int rt = 0; rt < 2; ++rt) {
      v8f a = {};
      a = gemm16(sEij, rt * 16, 128, sWm3, ob, 128, 128, a, lane);
      a = gemm16(sEji, rt * 16, 128, sWm4, ob, 128, 128, a, lane);
      float cc = sC1[n];
#pragma unroll
      for (int r = 0; r < 8; ++r) {
        int m = rt * 16 + r + lh * 8;
        float val = a[r] + cc + c2[(size_t)(b * NN + jb + m) * DD + n];
        sM[m * 128 + n] = f2bf(fmaxf(val, 0.0f));
      }
    }
    __syncthreads();
    // GEMM2: u_pre = m@W5^T + b5 + ef_ij
#pragma unroll
    for (int rt = 0; rt < 2; ++rt) {
      v8f a = {};
      a = gemm16(sM, rt * 16, 128, sW5, ob, 128, 128, a, lane);
      float bvv = b5[n];
#pragma unroll
      for (int r = 0; r < 8; ++r) {
        int m = rt * 16 + r + lh * 8;
        sU[m * 128 + n] = a[r] + bvv + sEF[m * 128 + n];
      }
    }
    __syncthreads();
    {   // LayerNorm 1 -> sU (f32) and sUb (bf16); 8 threads/row, 16 cols each
      int row = t >> 3, cb = (t & 7) << 4;
      float* p = sU + row * 128;
      float s = 0.f;
#pragma unroll
      for (int c = 0; c < 16; ++c) s += p[cb + c];
      s += __shfl_xor(s, 1); s += __shfl_xor(s, 2); s += __shfl_xor(s, 4);
      float mu = s * (1.0f / 128.0f);
      float v2 = 0.f;
#pragma unroll
      for (int c = 0; c < 16; ++c) { float d = p[cb + c] - mu; v2 += d * d; }
      v2 += __shfl_xor(v2, 1); v2 += __shfl_xor(v2, 2); v2 += __shfl_xor(v2, 4);
      float rstd = rsqrtf(v2 * (1.0f / 128.0f) + 1e-5f);
#pragma unroll
      for (int c = 0; c < 16; ++c) {
        int cc = cb + c;
        float y = (p[cc] - mu) * rstd * g1[cc] + bb1[cc];
        p[cc] = y;
        sUb[row * 128 + cc] = f2bf(y);
      }
    }
    __syncthreads();
    // GEMM3: r = relu(u@e1^T + e1_b)
#pragma unroll
    for (int rt = 0; rt < 2; ++rt) {
      v8f a = {};
      a = gemm16(sUb, rt * 16, 128, sE1, ob, 128, 128, a, lane);
      float bvv = e1b[n];
#pragma unroll
      for (int r = 0; r < 8; ++r) {
        int m = rt * 16 + r + lh * 8;
        sR[m * 128 + n] = f2bf(fmaxf(a[r] + bvv, 0.0f));
      }
    }
    __syncthreads();
    // GEMM4: h + u -> sEF
#pragma unroll
    for (int rt = 0; rt < 2; ++rt) {
      v8f a = {};
      a = gemm16(sR, rt * 16, 128, sE2, ob, 128, 128, a, lane);
      float bvv = e2b[n];
#pragma unroll
      for (int r = 0; r < 8; ++r) {
        int m = rt * 16 + r + lh * 8;
        sEF[m * 128 + n] = a[r] + bvv + sU[m * 128 + n];
      }
    }
    __syncthreads();
    {   // LayerNorm 2 -> global e
      int row = t >> 3, cb = (t & 7) << 4;
      float* p = sEF + row * 128;
      float s = 0.f;
#pragma unroll
      for (int c = 0; c < 16; ++c) s += p[cb + c];
      s += __shfl_xor(s, 1); s += __shfl_xor(s, 2); s += __shfl_xor(s, 4);
      float mu = s * (1.0f / 128.0f);
      float v2 = 0.f;
#pragma unroll
      for (int c = 0; c < 16; ++c) { float d = p[cb + c] - mu; v2 += d * d; }
      v2 += __shfl_xor(v2, 1); v2 += __shfl_xor(v2, 2); v2 += __shfl_xor(v2, 4);
      float rstd = rsqrtf(v2 * (1.0f / 128.0f) + 1e-5f);
      float* dst = eOut + ((size_t)bi * NN + jb + row) * DD;
#pragma unroll
      for (int c = 0; c < 16; ++c) {
        int cc = cb + c;
        dst[cc] = (p[cc] - mu) * rstd * g2[cc] + bb2[cc];
      }
    }
  }
}

// ---------------------------------------------------------------------------
extern "C" void kernel_launch(void* const* d_in, const int* in_sizes, int n_in,
                              void* d_out, int out_size, void* d_ws, size_t ws_size,
                              hipStream_t stream) {
  (void)in_sizes; (void)n_in; (void)out_size; (void)ws_size;
  const float* x     = (const float*)d_in[0];
  const float* ef    = (const float*)d_in[1];
  const float* Wqn_w = (const float*)d_in[2];
  const float* Wqn_b = (const float*)d_in[3];
  const float* Wkn_w = (const float*)d_in[4];
  const float* Wkn_b = (const float*)d_in[5];
  const float* Wvn_w = (const float*)d_in[6];
  const float* Wvn_b = (const float*)d_in[7];
  const float* Wqe_w = (const float*)d_in[8];
  const float* Wqe_b = (const float*)d_in[9];
  const float* Wke_w = (const float*)d_in[10];
  const float* Wke_b = (const float*)d_in[11];
  const float* Wve_w = (const float*)d_in[12];
  const float* Wve_b = (const float*)d_in[13];
  const float* m_w   = (const float*)d_in[14];
  const float* m_b   = (const float*)d_in[15];
  const float* W5_w  = (const float*)d_in[16];
  const float* W5_b  = (const float*)d_in[17];
  const float* ln1_g = (const float*)d_in[18];
  const float* ln1_b = (const float*)d_in[19];
  const float* e1_w  = (const float*)d_in[20];
  const float* e1_b  = (const float*)d_in[21];
  const float* e2_w  = (const float*)d_in[22];
  const float* e2_b  = (const float*)d_in[23];
  const float* ln2_g = (const float*)d_in[24];
  const float* ln2_b = (const float*)d_in[25];

  float* ws = (float*)d_ws;
  const size_t R = (size_t)BN * DD;   // 98304
  float* qn = ws;
  float* kn = qn + R;
  float* vn = kn + R;
  float* xn = vn + R;
  float* c1 = xn + R;
  float* c2 = c1 + R;

  float* xnOut = (float*)d_out;
  float* eOut  = xnOut + R;

  // node projections
  k_lin128<<<BN / 16, 256, 0, stream>>>(x, Wqn_w, 128, 0, Wqn_b, qn);
  k_lin128<<<BN / 16, 256, 0, stream>>>(x, Wkn_w, 128, 0, Wkn_b, kn);
  k_lin128<<<BN / 16, 256, 0, stream>>>(x, Wvn_w, 128, 0, Wvn_b, vn);
  // edge-conditioned attention -> xn
  k_attn<<<BN, 256, 0, stream>>>(ef, Wqe_w, Wqe_b, Wke_w, Wke_b, Wve_w, Wve_b,
                                 qn, kn, vn, xnOut, xn);
  // per-node halves of the concat MLP (m_w[:,0:128] and m_w[:,128:256])
  k_lin128<<<BN / 16, 256, 0, stream>>>(xn, m_w, 512, 0,   nullptr, c1);
  k_lin128<<<BN / 16, 256, 0, stream>>>(xn, m_w, 512, 128, nullptr, c2);
  // fused edge MLP + residual LayerNorms -> e
  k_edge<<<BN, 256, 0, stream>>>(ef, m_w, m_b, W5_w, W5_b, ln1_g, ln1_b,
                                 e1_w, e1_b, e2_w, e2_b, ln2_g, ln2_b,
                                 c1, c2, eOut);
}
